// PhysicsGraphFusion_69312182223261
// MI455X (gfx1250) — matrix operand order repaired and checked by reading, not compiled, not run on hardware
//
#include <hip/hip_runtime.h>
#include <hip/hip_bf16.h>

typedef __attribute__((ext_vector_type(16))) __bf16 v16bf;
typedef __attribute__((ext_vector_type(8)))  __bf16 v8bf;
typedef __attribute__((ext_vector_type(8)))  float  v8f;

#define BATCH 1024
#define NNODE 7
#define DDIM  1024
#define NROWS (BATCH * NNODE)   // 7168

__constant__ float ADJ_C[49] = {
    1,1,0,0,1,1,1,
    1,1,1,1,1,1,1,
    0,1,1,0,1,0,1,
    0,1,0,1,1,1,1,
    1,1,1,1,1,1,1,
    1,1,0,1,1,1,1,
    1,1,1,1,1,1,1};

__device__ __forceinline__ float gelu_f(float x) {
    // exact GELU: 0.5*x*(1+erf(x/sqrt(2)))
    return 0.5f * x * (1.0f + erff(x * 0.70710678118654752440f));
}

__device__ __forceinline__ float wave_reduce_add(float s) {
    #pragma unroll
    for (int off = 16; off; off >>= 1) s += __shfl_xor(s, off, 32);
    return s;
}

// ---------------------------------------------------------------------------
// Generic bf16 WMMA GEMM: C[M,N] = act( A[M,K] @ Bt[N,K]^T + bias + residual )
// Each wave computes a 16x64 tile: 1 A-fragment reused across 4 WMMAs/k-step.
// Fragment k order: contiguous k = 16*half + s for both A and B (wave32, 16-bit A 16x32, B 32x16).
// ---------------------------------------------------------------------------
__global__ __launch_bounds__(256)
void gemm_bf16_wmma(const __bf16* __restrict__ A, long lda,
                    const __bf16* __restrict__ Bt,       // N x K row-major
                    const float* __restrict__ bias,       // len N or null
                    const float* __restrict__ residual, long ldr,
                    float*  __restrict__ outF, long ldf,
                    __bf16* __restrict__ outB, long ldb,
                    int M, int N, int K, int act)
{
    const int wid  = threadIdx.x >> 5;
    const int lane = threadIdx.x & 31;
    const int half = lane >> 4;
    const int l    = lane & 15;

    const long tilesN4 = N >> 6;
    const long totalTiles = (long)(M >> 4) * tilesN4;
    long wgl = (long)blockIdx.x * (blockDim.x >> 5) + wid;
    if (wgl >= totalTiles) return;
    const int mt = (int)(wgl / tilesN4);
    const int nt = (int)(wgl % tilesN4);
    const int m0 = mt << 4;
    const int n0 = nt << 6;

    v8f acc0 = {}, acc1 = {}, acc2 = {}, acc3 = {};

    // A: row m0+l, K-elems {half*16..+16}  (matches B k order)
    const __bf16* arow = A + (long)(m0 + l) * lda;
    // B: row n0+l of Bt (column of B), K-elems {half*16..+16} (contiguous k, same as A)
    const __bf16* b0 = Bt + (long)(n0 + l)      * K + half * 16;
    const __bf16* b1 = Bt + (long)(n0 + 16 + l) * K + half * 16;
    const __bf16* b2 = Bt + (long)(n0 + 32 + l) * K + half * 16;
    const __bf16* b3 = Bt + (long)(n0 + 48 + l) * K + half * 16;

    for (int k0 = 0; k0 < K; k0 += 32) {
        v8bf alo = *(const v8bf*)(arow + k0 + half * 16);
        v8bf ahi = *(const v8bf*)(arow + k0 + half * 16 + 8);
        v16bf a;
        #pragma unroll
        for (int i = 0; i < 8; ++i) { a[i] = alo[i]; a[8 + i] = ahi[i]; }
        v16bf fb0 = *(const v16bf*)(b0 + k0);
        v16bf fb1 = *(const v16bf*)(b1 + k0);
        v16bf fb2 = *(const v16bf*)(b2 + k0);
        v16bf fb3 = *(const v16bf*)(b3 + k0);
        acc0 = __builtin_amdgcn_wmma_f32_16x16x32_bf16(false, a, false, fb0, (short)0, acc0, false, false);
        acc1 = __builtin_amdgcn_wmma_f32_16x16x32_bf16(false, a, false, fb1, (short)0, acc1, false, false);
        acc2 = __builtin_amdgcn_wmma_f32_16x16x32_bf16(false, a, false, fb2, (short)0, acc2, false, false);
        acc3 = __builtin_amdgcn_wmma_f32_16x16x32_bf16(false, a, false, fb3, (short)0, acc3, false, false);
    }

    // D layout: VGPR r -> row m0 + half*8 + r, col = n-subtile + l
    const int rbase = m0 + half * 8;
    #pragma unroll
    for (int r = 0; r < 8; ++r) {
        const int row = rbase + r;
        float vals[4] = { acc0[r], acc1[r], acc2[r], acc3[r] };
        #pragma unroll
        for (int j = 0; j < 4; ++j) {
            const int col = n0 + j * 16 + l;
            float x = vals[j];
            if (bias)     x += bias[col];
            if (residual) x += residual[(long)row * ldr + col];
            if (act == 1) x = gelu_f(x);
            if (outF) outF[(long)row * ldf + col] = x;
            if (outB) outB[(long)row * ldb + col] = (__bf16)x;
        }
    }
}

// ---------------------------------------------------------------------------
// fp32 -> bf16 with independent row strides (nodes -> concat buffer)
// ---------------------------------------------------------------------------
__global__ void conv_f32_bf16(const float* __restrict__ src, long src_ld,
                              __bf16* __restrict__ dst, long dst_ld,
                              long rows, long cols)
{
    long i = (long)blockIdx.x * blockDim.x + threadIdx.x;
    if (i >= rows * cols) return;
    long r = i / cols, c = i % cols;
    dst[r * dst_ld + c] = (__bf16)src[r * src_ld + c];
}

// fp32 (K x N row-major) -> bf16 transposed (N x K row-major)
__global__ void transpose_bf16(const float* __restrict__ src,
                               __bf16* __restrict__ dst, int K, int N)
{
    long i = (long)blockIdx.x * blockDim.x + threadIdx.x;
    if (i >= (long)N * K) return;
    long n = i / K, k = i % K;
    dst[i] = (__bf16)src[k * (long)N + n];
}

// ---------------------------------------------------------------------------
// Edge logits: e[b,n,m] = gelu(la[b,n]+rb[b,m]+cc[b]+be1) . We2 + be2
// One wave per (b,n,m).
// ---------------------------------------------------------------------------
__global__ __launch_bounds__(256)
void edge_kernel(const float* __restrict__ la, const float* __restrict__ rb,
                 const float* __restrict__ cc, const float* __restrict__ be1,
                 const float* __restrict__ We2, const float* __restrict__ be2,
                 float* __restrict__ edge_logits)
{
    const int wid = threadIdx.x >> 5, lane = threadIdx.x & 31;
    long w = (long)blockIdx.x * 8 + wid;
    if (w >= (long)BATCH * 49) return;
    const int b = (int)(w / 49), pmn = (int)(w % 49);
    const int n = pmn / 7, m = pmn % 7;
    const float* pa = la + ((long)b * 7 + n) * DDIM;
    const float* pb = rb + ((long)b * 7 + m) * DDIM;
    const float* pc = cc + (long)b * DDIM;
    float s = 0.f;
    for (int i = lane; i < DDIM; i += 32)
        s += gelu_f(pa[i] + pb[i] + pc[i] + be1[i]) * We2[i];
    s = wave_reduce_add(s);
    if (!lane) edge_logits[w] = s + be2[0];
}

// ---------------------------------------------------------------------------
// Attention per batch: qk^T/32 + edge + mask -> softmax -> messages (bf16
// into second half of upd_in); attn written to d_out.
// ---------------------------------------------------------------------------
__global__ __launch_bounds__(256)
void attn_kernel(const float* __restrict__ q, const float* __restrict__ k,
                 const float* __restrict__ v, const float* __restrict__ edge_logits,
                 float* __restrict__ attn_out, __bf16* __restrict__ upd_in)
{
    __shared__ float sL[49];
    __shared__ float sA[49];
    const int b = blockIdx.x, tid = threadIdx.x;
    const int wid = tid >> 5, lane = tid & 31;

    for (int p = wid; p < 49; p += 8) {
        const int n = p / 7, m = p % 7;
        const float* pq = q + ((long)b * 7 + n) * DDIM;
        const float* pk = k + ((long)b * 7 + m) * DDIM;
        float s = 0.f;
        for (int i = lane; i < DDIM; i += 32) s += pq[i] * pk[i];
        s = wave_reduce_add(s);
        if (!lane)
            sL[p] = s * (1.0f / 32.0f) + edge_logits[(long)b * 49 + p]
                    + (ADJ_C[p] - 1.0f) * 10000.0f;
    }
    __syncthreads();
    if (tid < 7) {
        const int n = tid;
        float mx = -1e30f;
        for (int m = 0; m < 7; ++m) mx = fmaxf(mx, sL[n * 7 + m]);
        float e[7], ss = 0.f;
        for (int m = 0; m < 7; ++m) { e[m] = __expf(sL[n * 7 + m] - mx); ss += e[m]; }
        for (int m = 0; m < 7; ++m) {
            float a = e[m] / ss;
            sA[n * 7 + m] = a;
            attn_out[(long)b * 49 + n * 7 + m] = a;
        }
    }
    __syncthreads();
    for (int d = tid; d < DDIM; d += blockDim.x) {
        float vv[7];
        #pragma unroll
        for (int m = 0; m < 7; ++m) vv[m] = v[((long)b * 7 + m) * DDIM + d];
        #pragma unroll
        for (int n = 0; n < 7; ++n) {
            float s = 0.f;
            #pragma unroll
            for (int m = 0; m < 7; ++m) s += sA[n * 7 + m] * vv[m];
            upd_in[((long)b * 7 + n) * 2048 + 1024 + d] = (__bf16)s;
        }
    }
}

// Broadcast updated[b, -1] (already bf16 in imp_in[:,:1024]) into imp_in[:,1024:]
__global__ void fill_gq(__bf16* __restrict__ imp_in)
{
    long i = (long)blockIdx.x * blockDim.x + threadIdx.x;
    if (i >= (long)NROWS * DDIM) return;
    long row = i / DDIM, d = i % DDIM;
    long b = row / 7;
    imp_in[row * 2048 + 1024 + d] = imp_in[(b * 7 + 6) * 2048 + d];
}

// imp_logits[row] = hidden_i[row] . Wi2 + bi2 ; one wave per row
__global__ __launch_bounds__(256)
void rowdot_kernel(const float* __restrict__ H, const float* __restrict__ w,
                   const float* __restrict__ bias, float* __restrict__ out)
{
    const int wid = threadIdx.x >> 5, lane = threadIdx.x & 31;
    long row = (long)blockIdx.x * 8 + wid;
    if (row >= NROWS) return;
    const float* h = H + row * DDIM;
    float s = 0.f;
    for (int i = lane; i < DDIM; i += 32) s += h[i] * w[i];
    s = wave_reduce_add(s);
    if (!lane) out[row] = s + bias[0];
}

// softmax + _cap_importance + fused = imp @ updated ; one block per batch
__global__ __launch_bounds__(256)
void importance_kernel(const float* __restrict__ imp_logits,
                       const float* __restrict__ updated,
                       float* __restrict__ imp_out, float* __restrict__ fused_out)
{
    __shared__ float sImp[7];
    const int b = blockIdx.x, tid = threadIdx.x;
    if (tid == 0) {
        float lg[7], mx = -1e30f;
        for (int n = 0; n < 7; ++n) { lg[n] = imp_logits[(long)b * 7 + n]; mx = fmaxf(mx, lg[n]); }
        float ss = 0.f;
        for (int n = 0; n < 7; ++n) { lg[n] = __expf(lg[n] - mx); ss += lg[n]; }
        float imp[7];
        for (int n = 0; n < 7; ++n) imp[n] = lg[n] / ss;
        const float cap[7] = {1.f,1.f,1.f,0.26f,1.f,1.f,0.24f};   // VIB_IDX=3, CTX_IDX=6
        const float fre[7] = {1.f,1.f,1.f,0.f,  1.f,1.f,0.f};
        float capped[7], capsum = 0.f, fmass = 0.f;
        for (int n = 0; n < 7; ++n) {
            capped[n] = fminf(imp[n], cap[n]);
            capsum += capped[n];
            fmass  += imp[n] * fre[n];
        }
        float resid = fmaxf(1.0f - capsum, 0.0f);
        float redis[7], redsum = 0.f;
        for (int n = 0; n < 7; ++n) {
            float fs = (fmass > 1e-6f) ? imp[n] * fre[n] / fmaxf(fmass, 1e-6f)
                                       : fre[n] * 0.2f;            // free/5
            redis[n] = capped[n] + fs * resid;
            redsum += redis[n];
        }
        float inv = 1.0f / fmaxf(redsum, 1e-6f);
        for (int n = 0; n < 7; ++n) {
            float r = redis[n] * inv;
            sImp[n] = r;
            imp_out[(long)b * 7 + n] = r;
        }
    }
    __syncthreads();
    for (int d = tid; d < DDIM; d += blockDim.x) {
        float s = 0.f;
        #pragma unroll
        for (int n = 0; n < 7; ++n) s += sImp[n] * updated[((long)b * 7 + n) * DDIM + d];
        fused_out[(long)b * DDIM + d] = s;
    }
}

// physics + alignment losses ; one block per batch ; 57 wave-level dots
__global__ __launch_bounds__(256)
void loss_kernel(const float* __restrict__ updated, const float* __restrict__ fused,
                 float* __restrict__ acc)
{
    __shared__ float sd[57];
    const int b = blockIdx.x, tid = threadIdx.x;
    const int wid = tid >> 5, lane = tid & 31;
    for (int t = wid; t < 57; t += 8) {
        const float *x, *y;
        if (t < 49) {           // pairwise dots (includes diag = norm^2)
            x = updated + ((long)b * 7 + t / 7) * DDIM;
            y = updated + ((long)b * 7 + t % 7) * DDIM;
        } else if (t < 56) {    // dot(updated[n], fused)
            x = updated + ((long)b * 7 + (t - 49)) * DDIM;
            y = fused + (long)b * DDIM;
        } else {                // ||fused||^2
            x = fused + (long)b * DDIM; y = x;
        }
        float s = 0.f;
        for (int i = lane; i < DDIM; i += 32) s += x[i] * y[i];
        s = wave_reduce_add(s);
        if (!lane) sd[t] = s;
    }
    __syncthreads();
    if (tid == 0) {
        float norms[7];
        for (int n = 0; n < 7; ++n) norms[n] = sqrtf(sd[n * 7 + n]);
        const float fn = fmaxf(sqrtf(sd[56]), 1e-12f);
        float e = 0.f, no = 0.f, al = 0.f;
        for (int n = 0; n < 7; ++n)
            for (int m = 0; m < 7; ++m) {
                float cs = sd[n * 7 + m] / fmaxf(norms[n] * norms[m], 1e-8f);
                e += (1.0f - cs) * ADJ_C[n * 7 + m];
                float nonm = (1.0f - ADJ_C[n * 7 + m]) * ((n == m) ? 0.0f : 1.0f);
                no += fmaxf(cs - 0.35f, 0.0f) * nonm;
            }
        for (int n = 0; n < 7; ++n)
            al += 1.0f - sd[49 + n] / (fmaxf(norms[n], 1e-12f) * fn);
        atomicAdd(&acc[0], e);
        atomicAdd(&acc[1], no);
        atomicAdd(&acc[2], al);
    }
}

__global__ void init_acc(float* acc) { if (threadIdx.x < 3) acc[threadIdx.x] = 0.f; }

__global__ void write_scalars(const float* __restrict__ acc, float* __restrict__ out)
{
    if (threadIdx.x == 0 && blockIdx.x == 0) {
        float phys  = acc[0] / 41.0f + 0.5f * acc[1] / 8.0f;  // ADJ.sum()=41, non_mask.sum()=8
        float align = acc[2] / (float)(BATCH * NNODE);
        out[0] = phys;   // phys + align*0.0
        out[1] = align;
    }
}

// ---------------------------------------------------------------------------
extern "C" void kernel_launch(void* const* d_in, const int* in_sizes, int n_in,
                              void* d_out, int out_size, void* d_ws, size_t ws_size,
                              hipStream_t stream)
{
    const float* nodes = (const float*)d_in[0];
    const float* Wq  = (const float*)d_in[1];  const float* bq  = (const float*)d_in[2];
    const float* Wk  = (const float*)d_in[3];  const float* bk  = (const float*)d_in[4];
    const float* Wv  = (const float*)d_in[5];  const float* bv  = (const float*)d_in[6];
    const float* We1 = (const float*)d_in[7];  const float* be1 = (const float*)d_in[8];
    const float* We2 = (const float*)d_in[9];  const float* be2 = (const float*)d_in[10];
    const float* Wu1 = (const float*)d_in[11]; const float* bu1 = (const float*)d_in[12];
    const float* Wu2 = (const float*)d_in[13]; const float* bu2 = (const float*)d_in[14];
    const float* Wi1 = (const float*)d_in[15]; const float* bi1 = (const float*)d_in[16];
    const float* Wi2 = (const float*)d_in[17]; const float* bi2 = (const float*)d_in[18];

    const long RD = (long)NROWS;       // 7168
    const int  D  = DDIM;

    // ---- workspace carve ----
    char* p = (char*)d_ws;
    auto carve = [&](size_t bytes) -> void* {
        void* r = (void*)p; p += (bytes + 255) & ~(size_t)255; return r;
    };
    __bf16* upd_in = (__bf16*)carve(RD * 2048 * 2);   // [nodes | messages] bf16
    __bf16* imp_in = (__bf16*)carve(RD * 2048 * 2);   // [updated | gq] bf16
    __bf16* hid_u  = (__bf16*)carve(RD * 1024 * 2);   // gelu(upd_in@Wu1+bu1) bf16
    __bf16* WqT  = (__bf16*)carve((size_t)D * D * 2);
    __bf16* WkT  = (__bf16*)carve((size_t)D * D * 2);
    __bf16* WvT  = (__bf16*)carve((size_t)D * D * 2);
    __bf16* WaT  = (__bf16*)carve((size_t)D * D * 2);
    __bf16* WbT  = (__bf16*)carve((size_t)D * D * 2);
    __bf16* WcT  = (__bf16*)carve((size_t)D * D * 2);
    __bf16* Wu2T = (__bf16*)carve((size_t)D * D * 2);
    __bf16* Wu1T = (__bf16*)carve((size_t)D * 2048 * 2);
    __bf16* Wi1T = (__bf16*)carve((size_t)D * 2048 * 2);
    float* qbuf  = (float*)carve(RD * 1024 * 4);
    float* kbuf  = (float*)carve(RD * 1024 * 4);
    float* vbuf  = (float*)carve(RD * 1024 * 4);
    float* labuf = (float*)carve(RD * 1024 * 4);
    float* rbbuf = (float*)carve(RD * 1024 * 4);
    float* ccbuf = (float*)carve((size_t)BATCH * 1024 * 4);
    float* hid_i = (float*)carve(RD * 1024 * 4);
    float* edgeL = (float*)carve((size_t)BATCH * 49 * 4);
    float* impL  = (float*)carve(RD * 4);
    float* acc   = (float*)carve(256);

    // ---- d_out layout: fused | updated | imp | attn | phys | align ----
    float* outF        = (float*)d_out;
    float* out_fused   = outF;
    float* out_updated = outF + (long)BATCH * D;
    float* out_imp     = out_updated + RD * D;
    float* out_attn    = out_imp + (long)BATCH * NNODE;
    float* out_scal    = out_attn + (long)BATCH * 49;

    auto cdiv = [](long a, long b) { return (int)((a + b - 1) / b); };
    auto gemm = [&](const __bf16* A, long lda, const __bf16* Bt,
                    const float* bias, const float* res, long ldr,
                    float* oF, long ldf, __bf16* oB, long ldb,
                    int M, int N, int K, int act) {
        long waves = ((long)M / 16) * ((long)N / 64);
        gemm_bf16_wmma<<<cdiv(waves, 8), 256, 0, stream>>>(
            A, lda, Bt, bias, res, ldr, oF, ldf, oB, ldb, M, N, K, act);
    };

    init_acc<<<1, 64, 0, stream>>>(acc);

    // nodes fp32 -> bf16 into first half of upd_in (row stride 2048)
    conv_f32_bf16<<<cdiv(RD * D, 256), 256, 0, stream>>>(nodes, D, upd_in, 2048, RD, D);

    // weights -> bf16, transposed to N x K
    const long TW = (long)D * D;
    transpose_bf16<<<cdiv(TW, 256), 256, 0, stream>>>(Wq,  WqT,  D, D);
    transpose_bf16<<<cdiv(TW, 256), 256, 0, stream>>>(Wk,  WkT,  D, D);
    transpose_bf16<<<cdiv(TW, 256), 256, 0, stream>>>(Wv,  WvT,  D, D);
    transpose_bf16<<<cdiv(TW, 256), 256, 0, stream>>>(We1,            WaT, D, D);
    transpose_bf16<<<cdiv(TW, 256), 256, 0, stream>>>(We1 + TW,       WbT, D, D);
    transpose_bf16<<<cdiv(TW, 256), 256, 0, stream>>>(We1 + 2 * TW,   WcT, D, D);
    transpose_bf16<<<cdiv(TW, 256), 256, 0, stream>>>(Wu2, Wu2T, D, D);
    transpose_bf16<<<cdiv(2 * TW, 256), 256, 0, stream>>>(Wu1, Wu1T, 2048, D);
    transpose_bf16<<<cdiv(2 * TW, 256), 256, 0, stream>>>(Wi1, Wi1T, 2048, D);

    // q/k/v = nodes@W + b ; la/rb = nodes@Wa/Wb ; cc = nodes[:,-1]@Wc
    gemm(upd_in, 2048, WqT, bq, nullptr, 0, qbuf,  D, nullptr, 0, NROWS, D, D, 0);
    gemm(upd_in, 2048, WkT, bk, nullptr, 0, kbuf,  D, nullptr, 0, NROWS, D, D, 0);
    gemm(upd_in, 2048, WvT, bv, nullptr, 0, vbuf,  D, nullptr, 0, NROWS, D, D, 0);
    gemm(upd_in, 2048, WaT, nullptr, nullptr, 0, labuf, D, nullptr, 0, NROWS, D, D, 0);
    gemm(upd_in, 2048, WbT, nullptr, nullptr, 0, rbbuf, D, nullptr, 0, NROWS, D, D, 0);
    gemm(upd_in + 6 * 2048, 7 * 2048, WcT, nullptr, nullptr, 0,
         ccbuf, D, nullptr, 0, BATCH, D, D, 0);

    // edge logits, attention + messages (messages -> upd_in[:,1024:])
    edge_kernel<<<(BATCH * 49) / 8, 256, 0, stream>>>(labuf, rbbuf, ccbuf, be1, We2, be2, edgeL);
    attn_kernel<<<BATCH, 256, 0, stream>>>(qbuf, kbuf, vbuf, edgeL, out_attn, upd_in);

    // update MLP: hid_u = gelu(upd_in@Wu1+bu1); updated = hid_u@Wu2+bu2+nodes
    gemm(upd_in, 2048, Wu1T, bu1, nullptr, 0,
         nullptr, 0, hid_u, 1024, NROWS, D, 2048, 1);
    gemm(hid_u, 1024, Wu2T, bu2, nodes, D,
         out_updated, D, imp_in, 2048, NROWS, D, D, 0);   // bf16 copy -> imp_in[:, :1024]

    // importance MLP: gq broadcast, hid_i = gelu(imp_in@Wi1+bi1), dot Wi2
    fill_gq<<<cdiv(RD * D, 256), 256, 0, stream>>>(imp_in);
    gemm(imp_in, 2048, Wi1T, bi1, nullptr, 0,
         hid_i, D, nullptr, 0, NROWS, D, 2048, 1);
    rowdot_kernel<<<cdiv(RD, 8), 256, 0, stream>>>(hid_i, Wi2, bi2, impL);

    // softmax + cap + fused ; losses ; scalars
    importance_kernel<<<BATCH, 256, 0, stream>>>(impL, out_updated, out_imp, out_fused);
    loss_kernel<<<BATCH, 256, 0, stream>>>(out_updated, out_fused, acc);
    write_scalars<<<1, 64, 0, stream>>>(acc, out_scal);
}
